// E63mMatrixNonlinear_4887672783114
// MI455X (gfx1250) — matrix operand order repaired and checked by reading, not compiled, not run on hardware
//
#include <hip/hip_runtime.h>
#include <hip/hip_bf16.h>

// ---------------------------------------------------------------------------
// Problem constants (from the reference): T=512, B=8, D=512, N=64
// ---------------------------------------------------------------------------
#define TT   512
#define BB   8
#define DD   512
#define NN   64
#define KBLKS (DD / 32)          // 16 WMMA K-steps of 32 over K=512
#define NSUB  4                  // N-tiles per wave strip (16x64 output/wave)

typedef __attribute__((ext_vector_type(16))) __bf16 v16bf;
typedef __attribute__((ext_vector_type(8)))  float  v8f;
typedef __attribute__((ext_vector_type(4)))  float  v4f;   // native vector for NT stores

// ---------------------------------------------------------------------------
// Pack A fragments: x [M,K] fp32 -> bf16, laid out per-lane in the CDNA5
// 16-bit A-matrix (16x32) VGPR order so the GEMM does one contiguous
// 32-byte load per fragment.
//   lane l: h=l>>4, m=tile_m*16+(l&15);  element j (0..15): K = 16*(j>>3) + 8*h + (j&7)
// Buffer layout: [Mtiles][KBLKS][32 lanes][16 bf16]
// ---------------------------------------------------------------------------
__global__ void pack_a_kernel(const float* __restrict__ x, __bf16* __restrict__ ap,
                              int Mtiles, int K) {
    int gid = blockIdx.x * blockDim.x + threadIdx.x;
    int total = Mtiles * KBLKS * 32;
    if (gid >= total) return;
    int lane = gid & 31;
    int rest = gid >> 5;
    int kblk = rest % KBLKS;
    int mt   = rest / KBLKS;
    int h = lane >> 4;
    int m = mt * 16 + (lane & 15);
    const float* src = x + (size_t)m * K + kblk * 32;
    v16bf tmp;
#pragma unroll
    for (int j = 0; j < 16; ++j) {
        int kk = 16 * (j >> 3) + 8 * h + (j & 7);
        tmp[j] = (__bf16)src[kk];
    }
    *(v16bf*)(ap + (size_t)gid * 16) = tmp;
}

// ---------------------------------------------------------------------------
// Pack B fragments: W [Nout,K] row-major fp32 -> bf16 in B-matrix (32x16)
// per-lane order. B[k][n] = W[n][k].
//   lane l: h=l>>4, n=tile_n*16+(l&15); element j: K = 16*(j>>3) + 8*h + (j&7)
// Buffer layout: [Ntiles][KBLKS][32 lanes][16 bf16]
// ---------------------------------------------------------------------------
__global__ void pack_b_kernel(const float* __restrict__ w, __bf16* __restrict__ bp,
                              int Ntiles, int K) {
    int gid = blockIdx.x * blockDim.x + threadIdx.x;
    int total = Ntiles * KBLKS * 32;
    if (gid >= total) return;
    int lane = gid & 31;
    int rest = gid >> 5;
    int kblk = rest % KBLKS;
    int nt   = rest / KBLKS;
    int h = lane >> 4;
    int n = nt * 16 + (lane & 15);
    const float* src = w + (size_t)n * K + kblk * 32;
    v16bf tmp;
#pragma unroll
    for (int j = 0; j < 16; ++j) tmp[j] = (__bf16)src[16 * (j >> 3) + 8 * h + (j & 7)];
    *(v16bf*)(bp + (size_t)gid * 16) = tmp;
}

// ---------------------------------------------------------------------------
// GEMM: out[M,Nout] = X[M,K] * W^T, fp32 accumulate via
// v_wmma_f32_16x16x32_bf16. Each wave owns a 16x64 output strip (4 N-tiles):
// the A fragment is loaded once per K-step and reused by 4 WMMAs -> 4x the
// arithmetic intensity of a tile-per-wave scheme, 64 WMMAs per strip.
// 256 threads = 8 waves per block.
// ---------------------------------------------------------------------------
__global__ void gemm_bf16_kernel(const __bf16* __restrict__ ap,
                                 const __bf16* __restrict__ bp,
                                 float* __restrict__ out,
                                 int Mtiles, int Ntiles, int Nout) {
    int wave = threadIdx.x >> 5;
    int lane = threadIdx.x & 31;
    int Nstrips = Ntiles / NSUB;
    int strip = blockIdx.x * (blockDim.x >> 5) + wave;
    if (strip >= Mtiles * Nstrips) return;
    int tm  = strip / Nstrips;
    int tn0 = (strip % Nstrips) * NSUB;

    const v16bf* A  = (const v16bf*)ap + (size_t)tm * KBLKS * 32 + lane;
    const v16bf* B0 = (const v16bf*)bp + (size_t)(tn0 + 0) * KBLKS * 32 + lane;
    const v16bf* B1 = (const v16bf*)bp + (size_t)(tn0 + 1) * KBLKS * 32 + lane;
    const v16bf* B2 = (const v16bf*)bp + (size_t)(tn0 + 2) * KBLKS * 32 + lane;
    const v16bf* B3 = (const v16bf*)bp + (size_t)(tn0 + 3) * KBLKS * 32 + lane;

    v8f c0 = {}, c1 = {}, c2 = {}, c3 = {};
#pragma unroll
    for (int kb = 0; kb < KBLKS; ++kb) {
        if (kb + 2 < KBLKS) {
            __builtin_prefetch(&A[(kb + 2) * 32], 0, 1);   // global_prefetch
            __builtin_prefetch(&B0[(kb + 2) * 32], 0, 1);
            __builtin_prefetch(&B1[(kb + 2) * 32], 0, 1);
            __builtin_prefetch(&B2[(kb + 2) * 32], 0, 1);
            __builtin_prefetch(&B3[(kb + 2) * 32], 0, 1);
        }
        v16bf a  = A[kb * 32];
        v16bf b0 = B0[kb * 32];
        v16bf b1 = B1[kb * 32];
        v16bf b2 = B2[kb * 32];
        v16bf b3 = B3[kb * 32];
        // (neg_a, A, neg_b, B, c_mod, C, reuse_a, reuse_b)
        c0 = __builtin_amdgcn_wmma_f32_16x16x32_bf16(false, a, false, b0,
                                                     (short)0, c0, false, false);
        c1 = __builtin_amdgcn_wmma_f32_16x16x32_bf16(false, a, false, b1,
                                                     (short)0, c1, false, false);
        c2 = __builtin_amdgcn_wmma_f32_16x16x32_bf16(false, a, false, b2,
                                                     (short)0, c2, false, false);
        c3 = __builtin_amdgcn_wmma_f32_16x16x32_bf16(false, a, false, b3,
                                                     (short)0, c3, false, false);
    }

    // C/D layout: VGPR r -> M = r + 8*(lane>>4), N = lane&15
    int h  = lane >> 4;
    int nn = lane & 15;
    float* orow = out + (size_t)(tm * 16 + 8 * h) * Nout + nn;
#pragma unroll
    for (int r = 0; r < 8; ++r) {
        size_t ro = (size_t)r * Nout;
        orow[ro + (tn0 + 0) * 16] = c0[r];
        orow[ro + (tn0 + 1) * 16] = c1[r];
        orow[ro + (tn0 + 2) * 16] = c2[r];
        orow[ro + (tn0 + 3) * 16] = c3[r];
    }
}

// ---------------------------------------------------------------------------
// Recurrent scan. One block per batch b (8 blocks, 512 threads = 16 waves).
// State S_b [N=64, D=512] = 128 KB lives in LDS for all T steps (WGP has
// 320 KB). Per step:
//   Sk = S @ k      (LDS float4 dot + ds_add_f32 reduction)
//   v  = tanh(tanh(Sk) @ W_r^T + wx + b)
//   alpha = sigmoid(ax + b_alpha)
//   S  = alpha*S + (1-alpha) * v (outer) k     (update in LDS, stream to HBM NT)
//   y  = tanh(S @ q)
// ---------------------------------------------------------------------------
__global__ void scan_kernel(const float* __restrict__ S0,
                            const float* __restrict__ Wr,
                            const float* __restrict__ bvec,
                            const float* __restrict__ balpha,
                            const float* __restrict__ k_all,
                            const float* __restrict__ q_all,
                            const float* __restrict__ wx_all,
                            const float* __restrict__ ax_all,
                            float* __restrict__ y_out,
                            float* __restrict__ S_out) {
    extern __shared__ float smem[];
    float* S   = smem;              // N*D = 32768
    float* WrL = S   + NN * DD;     // 4096
    float* kL  = WrL + NN * NN;     // 512
    float* qL  = kL  + DD;          // 512
    float* Sk  = qL  + DD;          // 64
    float* ret = Sk  + NN;          // 64
    float* vL  = ret + NN;          // 64
    float* aL  = vL  + NN;          // 64

    const int tid = threadIdx.x;    // 512 threads
    const int b   = blockIdx.x;     // 0..7

    // init: S <- S0, emit S_out[t=0] slab, cache W_r in LDS
    {
        const float* s0  = S0 + (size_t)b * NN * DD;
        float*       so0 = S_out + (size_t)b * NN * DD;   // ((0)*BB + b) slab
        for (int i = tid; i < NN * DD; i += blockDim.x) {
            float v = s0[i];
            S[i] = v;
            so0[i] = v;
        }
        for (int i = tid; i < NN * NN; i += blockDim.x) WrL[i] = Wr[i];
    }
    __syncthreads();

    for (int t = 0; t < TT; ++t) {
        const size_t base = (size_t)(t * BB + b);
        kL[tid] = k_all[base * DD + tid];
        qL[tid] = q_all[base * DD + tid];
        if (tid < NN) Sk[tid] = 0.0f;
        if (t + 1 < TT) {
            __builtin_prefetch(&k_all[(base + BB) * DD + tid], 0, 1);
            __builtin_prefetch(&q_all[(base + BB) * DD + tid], 0, 1);
        }
        __syncthreads();

        // ---- Sk[n] = sum_d S[n][d] * k[d] : 8 threads per row ----
        {
            int n = tid >> 3, sub = tid & 7;
            const float4* Sr = (const float4*)(S + n * DD);
            const float4* k4 = (const float4*)kL;
            float p = 0.0f;
#pragma unroll
            for (int i = 0; i < 16; ++i) {
                float4 s4 = Sr[sub * 16 + i];
                float4 kk = k4[sub * 16 + i];
                p += s4.x * kk.x + s4.y * kk.y + s4.z * kk.z + s4.w * kk.w;
            }
            atomicAdd(&Sk[n], p);   // ds_add_f32
        }
        __syncthreads();

        if (tid < NN) ret[tid] = tanhf(Sk[tid]);
        __syncthreads();

        // ---- v, alpha (one thread per n) ----
        if (tid < NN) {
            float acc = wx_all[base * NN + tid] + bvec[tid];
#pragma unroll 8
            for (int m = 0; m < NN; ++m) acc += WrL[tid * NN + m] * ret[m];
            vL[tid] = tanhf(acc);
            float ax = ax_all[base * NN + tid] + balpha[tid];
            aL[tid]  = 1.0f / (1.0f + expf(-ax));
            Sk[tid]  = 0.0f;        // reuse for the y reduction
        }
        __syncthreads();

        // ---- gated update + stream S_new to HBM (nontemporal) ----
        {
            float4*       S4  = (float4*)S;
            const float4* k4  = (const float4*)kL;
            v4f*          So4 = (v4f*)(S_out + ((size_t)(t + 1) * BB + b) * NN * DD);
            int d4    = tid & 127;      // float4 column
            int nbase = tid >> 7;       // 0..3
            float4 kk = k4[d4];
#pragma unroll
            for (int it = 0; it < 16; ++it) {
                int n = it * 4 + nbase;
                float al = aL[n], vv = vL[n];
                float om = (1.0f - al) * vv;
                int idx = n * 128 + d4;
                float4 s = S4[idx];
                s.x = al * s.x + om * kk.x;
                s.y = al * s.y + om * kk.y;
                s.z = al * s.z + om * kk.z;
                s.w = al * s.w + om * kk.w;
                S4[idx] = s;
                v4f sv = { s.x, s.y, s.z, s.w };
                __builtin_nontemporal_store(sv, &So4[idx]);  // write-once stream
            }
        }
        __syncthreads();

        // ---- y[n] = tanh(sum_d S_new[n][d] * q[d]) ----
        {
            int n = tid >> 3, sub = tid & 7;
            const float4* Sr = (const float4*)(S + n * DD);
            const float4* q4 = (const float4*)qL;
            float p = 0.0f;
#pragma unroll
            for (int i = 0; i < 16; ++i) {
                float4 s4 = Sr[sub * 16 + i];
                float4 qq = q4[sub * 16 + i];
                p += s4.x * qq.x + s4.y * qq.y + s4.z * qq.z + s4.w * qq.w;
            }
            atomicAdd(&Sk[n], p);
        }
        __syncthreads();

        if (tid < NN) y_out[base * NN + tid] = tanhf(Sk[tid]);
        __syncthreads();   // protect kL/qL/Sk before next step overwrites
    }
}

// ---------------------------------------------------------------------------
// Launcher
// ---------------------------------------------------------------------------
extern "C" void kernel_launch(void* const* d_in, const int* in_sizes, int n_in,
                              void* d_out, int out_size, void* d_ws, size_t ws_size,
                              hipStream_t stream) {
    (void)in_sizes; (void)n_in; (void)out_size; (void)ws_size;

    const float* x   = (const float*)d_in[0];   // [T,B,D]
    const float* S0  = (const float*)d_in[1];   // [B,N,D]
    const float* W_k = (const float*)d_in[2];   // [D,D]
    const float* W_q = (const float*)d_in[3];   // [D,D]
    const float* W_x = (const float*)d_in[4];   // [N,D]
    const float* W_r = (const float*)d_in[5];   // [N,N]
    const float* bv  = (const float*)d_in[6];   // [N]
    const float* W_a = (const float*)d_in[7];   // [N,D]
    const float* b_a = (const float*)d_in[8];   // [N]

    const int M      = TT * BB;     // 4096 rows
    const int Mtiles = M / 16;      // 256
    const int NtD    = DD / 16;     // 32  (Nout = 512)
    const int NtN    = NN / 16;     // 4   (Nout = 64)

    // workspace carve-up (bytes)
    char* ws = (char*)d_ws;
    size_t off = 0;
    __bf16* XP  = (__bf16*)(ws + off); off += (size_t)M  * DD * 2;   // 4 MB
    __bf16* WKP = (__bf16*)(ws + off); off += (size_t)DD * DD * 2;   // 512 KB
    __bf16* WQP = (__bf16*)(ws + off); off += (size_t)DD * DD * 2;
    __bf16* WXP = (__bf16*)(ws + off); off += (size_t)NN * DD * 2;   // 64 KB
    __bf16* WAP = (__bf16*)(ws + off); off += (size_t)NN * DD * 2;
    float* k_all  = (float*)(ws + off); off += (size_t)M * DD * 4;   // 8 MB
    float* q_all  = (float*)(ws + off); off += (size_t)M * DD * 4;
    float* wx_all = (float*)(ws + off); off += (size_t)M * NN * 4;   // 1 MB
    float* ax_all = (float*)(ws + off); off += (size_t)M * NN * 4;

    // ---- pack operands to bf16 WMMA fragment order ----
    {
        int tot = Mtiles * KBLKS * 32;
        pack_a_kernel<<<(tot + 255) / 256, 256, 0, stream>>>(x, XP, Mtiles, DD);
    }
    {
        int totD = NtD * KBLKS * 32;
        pack_b_kernel<<<(totD + 255) / 256, 256, 0, stream>>>(W_k, WKP, NtD, DD);
        pack_b_kernel<<<(totD + 255) / 256, 256, 0, stream>>>(W_q, WQP, NtD, DD);
        int totN = NtN * KBLKS * 32;
        pack_b_kernel<<<(totN + 255) / 256, 256, 0, stream>>>(W_x, WXP, NtN, DD);
        pack_b_kernel<<<(totN + 255) / 256, 256, 0, stream>>>(W_a, WAP, NtN, DD);
    }

    // ---- projection GEMMs (WMMA bf16 -> f32), 16x64 strip per wave ----
    {
        int stripsD  = Mtiles * (NtD / NSUB);       // 256*8 = 2048
        int blocksD  = (stripsD + 7) / 8;           // 8 waves per block
        gemm_bf16_kernel<<<blocksD, 256, 0, stream>>>(XP, WKP, k_all, Mtiles, NtD, DD);
        gemm_bf16_kernel<<<blocksD, 256, 0, stream>>>(XP, WQP, q_all, Mtiles, NtD, DD);
        int stripsN  = Mtiles * (NtN / NSUB);       // 256
        int blocksN  = (stripsN + 7) / 8;
        gemm_bf16_kernel<<<blocksN, 256, 0, stream>>>(XP, WXP, wx_all, Mtiles, NtN, NN);
        gemm_bf16_kernel<<<blocksN, 256, 0, stream>>>(XP, WAP, ax_all, Mtiles, NtN, NN);
    }

    // ---- recurrent scan: one block per batch, S resident in LDS ----
    {
        float* y_out = (float*)d_out;                         // [T,B,N]
        float* S_out = (float*)d_out + (size_t)TT * BB * NN;  // [T+1,B,N,D]
        size_t smem = (size_t)(NN * DD + NN * NN + 2 * DD + 4 * NN) * sizeof(float);
        scan_kernel<<<BB, DD, smem, stream>>>(S0, W_r, bv, b_a,
                                              k_all, q_all, wx_all, ax_all,
                                              y_out, S_out);
    }
}
